// AlineSS_87840671137862
// MI455X (gfx1250) — matrix operand order repaired and checked by reading, not compiled, not run on hardware
//
#include <hip/hip_runtime.h>

#define B_   2
#define H_   128
#define W_   128
#define DM_  96
#define DE_  192
#define NS_  16
#define RR_  6
#define MTOT (B_*H_*W_)   // 32768 pixels

typedef __attribute__((ext_vector_type(16))) _Float16 v16h;
typedef __attribute__((ext_vector_type(8)))  _Float16 v8h;
typedef __attribute__((ext_vector_type(8)))  float    v8f;

__device__ __forceinline__ float sigmoidf_(float x){ return 1.f/(1.f+__expf(-x)); }

// ---------------- fp32 -> f16 convert ----------------
__global__ __launch_bounds__(256)
void k_cvt_f16(const float* __restrict__ src, _Float16* __restrict__ dst, int n){
  int i = blockIdx.x*blockDim.x + threadIdx.x;
  if (i < n) dst[i] = (_Float16)src[i];
}

// pack x_proj_weight (K=2, 38, 192) into padded (80, 192) f16: rows [k*40, k*40+38)
__global__ __launch_bounds__(256)
void k_pack_xproj(const float* __restrict__ xw, _Float16* __restrict__ dst){
  int i = blockIdx.x*blockDim.x + threadIdx.x;     // 80*192
  if (i >= 80*DE_) return;
  int r = i / DE_, d = i % DE_;
  int k = r / 40, c = r % 40;
  float v = (c < (RR_ + 2*NS_)) ? xw[(k*(RR_+2*NS_) + c)*DE_ + d] : 0.f;
  dst[i] = (_Float16)v;
}

// ---------------- WMMA GEMM: C[M,N] = A[M,K](f16) * Wt[N,K]^T(f16), f32 accum --------
// grid (N/16, M/64), 128 threads = 4 waves; wave wv owns M-tile blockIdx.y*4+wv,
// all 4 waves share one N-tile whose B fragment is staged in LDS via async copy.
// K multiple of 32 (<=192), M multiple of 64, N multiple of 16.
__global__ __launch_bounds__(128)
void k_wmma_gemm(const _Float16* __restrict__ A, const _Float16* __restrict__ Wt,
                 float* __restrict__ C, int M, int N, int K){
  __shared__ _Float16 sB[16*DE_];                 // 16 rows x K (max 192) f16 = 6 KB
  int tid  = threadIdx.x;
  int lane = tid & 31;
  int wv   = tid >> 5;
  int tn = blockIdx.x, tm = blockIdx.y*4 + wv;
  bool hi = lane >= 16;
  int l15 = lane & 15;

  // ---- async-copy the 16xK B tile (contiguous rows of Wt) into LDS ----
  {
    const char* gbase = (const char*)(Wt + (size_t)tn * 16 * K);
    unsigned lbase = (unsigned)(unsigned long long)(&sB[0]);   // LDS offset (addr[31:0])
    int nchunks = 2*K;                                         // 16*K*2 bytes / 16B
    for (int c = tid; c < nchunks; c += 128) {
      unsigned long long ga = (unsigned long long)(gbase + c*16);
      unsigned la = lbase + c*16;
      asm volatile("global_load_async_to_lds_b128 %0, %1, off"
                   :: "v"(la), "v"(ga) : "memory");
    }
    asm volatile("s_wait_asynccnt 0x0" ::: "memory");
    __syncthreads();
  }

  const _Float16* arow  = A  + (size_t)(tm*16 + l15) * K;
  const _Float16* sBrow = sB + l15*K + (hi ? 16 : 0);
  int base = hi ? 16 : 0;  // A lanes 0-15 hold K {0..15}; lanes 16-31 hold {16..31}
  v8f acc = {};
  for (int k0 = 0; k0 < K; k0 += 32) {
    v8h c0 = *(const v8h*)(arow + k0 + base);
    v8h c1 = *(const v8h*)(arow + k0 + 8 + base);
    v16h a;
    #pragma unroll
    for (int j = 0; j < 8; ++j) { a[j] = c0[j]; a[8+j] = c1[j]; }
    v16h b = *(const v16h*)(sBrow + k0);          // 2x ds_load_b128
    acc = __builtin_amdgcn_wmma_f32_16x16x32_f16(false, a, false, b, (short)0, acc, false, false);
  }
  int orow = tm*16 + (hi ? 8 : 0);
  int ocol = tn*16 + l15;
  #pragma unroll
  for (int r = 0; r < 8; ++r)
    C[(size_t)(orow + r)*N + ocol] = acc[r];
}

// ---------------- depthwise 3x3 conv + bias + SiLU; writes fp32 (scan) + f16 (GEMM) ----
__global__ __launch_bounds__(256)
void k_conv_dw(const float* __restrict__ xz, const float* __restrict__ cw,
               const float* __restrict__ cb, float* __restrict__ xc,
               _Float16* __restrict__ xch){
  int i = blockIdx.x*blockDim.x + threadIdx.x;
  if (i >= MTOT*DE_) return;
  int d = i % DE_;
  int m = i / DE_;
  int w = m % W_; int h = (m / W_) % H_; int b = m / (H_*W_);
  float s = cb[d];
  #pragma unroll
  for (int ky = 0; ky < 3; ++ky){
    int hh = h + ky - 1;
    if (hh < 0 || hh >= H_) continue;
    #pragma unroll
    for (int kx = 0; kx < 3; ++kx){
      int ww = w + kx - 1;
      if (ww < 0 || ww >= W_) continue;
      s += xz[((size_t)((b*H_+hh)*W_+ww))*(2*DE_) + d] * cw[d*9 + ky*3 + kx];
    }
  }
  float r = s * sigmoidf_(s);
  xc[i]  = r;
  xch[i] = (_Float16)r;
}

// ---------------- selective scan: block = (sequence bw, direction k), thread = channel d
__global__ __launch_bounds__(192)
void k_scan(const float* __restrict__ xc, const float* __restrict__ proj,
            const float* __restrict__ dtw, const float* __restrict__ dtb,
            const float* __restrict__ alog, const float* __restrict__ Dsv,
            float* __restrict__ ys){
  int bwk = blockIdx.x;            // bw*2 + k
  int k  = bwk & 1;
  int bw = bwk >> 1;
  int b  = bw / W_;
  int w  = bw % W_;
  int d  = threadIdx.x;
  int kd = k*DE_ + d;
  float Arow[NS_];
  #pragma unroll
  for (int n = 0; n < NS_; ++n) Arow[n] = -__expf(alog[kd*NS_ + n]);
  float wdt[RR_];
  #pragma unroll
  for (int r = 0; r < RR_; ++r) wdt[r] = dtw[(size_t)kd*RR_ + r];
  float bias = dtb[kd];
  float Dv   = Dsv[kd];
  float hst[NS_];
  #pragma unroll
  for (int n = 0; n < NS_; ++n) hst[n] = 0.f;
  size_t ybase = (size_t)bwk * H_ * DE_;
  for (int l = 0; l < H_; ++l) {
    int hh = k ? (H_-1-l) : l;
    int m  = (b*H_ + hh)*W_ + w;
    float uu = xc[(size_t)m*DE_ + d];
    const float* pr = proj + (size_t)m*80 + k*40;   // [dt(6) | B(16) | C(16) | pad]
    if (l + 1 < H_) {
      int hh2 = k ? (H_-2-l) : (l+1);
      __builtin_prefetch(xc + (size_t)((b*H_+hh2)*W_+w)*DE_ + d, 0, 1);
    }
    float dtv = bias;
    #pragma unroll
    for (int r = 0; r < RR_; ++r) dtv += wdt[r]*pr[r];
    float delta = (dtv > 20.f) ? dtv : __logf(1.f + __expf(dtv));   // softplus
    float du = delta * uu;
    const float* Bl = pr + RR_;
    const float* Cl = pr + RR_ + NS_;
    float y = 0.f;
    #pragma unroll
    for (int n = 0; n < NS_; ++n) {
      float dA = __expf(delta * Arow[n]);
      hst[n] = dA*hst[n] + du*Bl[n];
      y += hst[n]*Cl[n];
    }
    ys[ybase + (size_t)l*DE_ + d] = y + Dv*uu;
  }
}

// ---------------- merge fwd/bwd + LayerNorm + SiLU(z) gate -> f16 for out_proj ---------
__global__ __launch_bounds__(192)
void k_merge_ln(const float* __restrict__ ys, const float* __restrict__ xz,
                const float* __restrict__ gam, const float* __restrict__ bet,
                _Float16* __restrict__ ylnh){
  int m = blockIdx.x;              // (b*H+h)*W + w
  int d = threadIdx.x;
  int w = m % W_; int h = (m / W_) % H_; int b = m / (H_*W_);
  int bw = b*W_ + w;
  float v = ys[((size_t)(bw*2+0)*H_ + h)*DE_ + d]
          + ys[((size_t)(bw*2+1)*H_ + (H_-1-h))*DE_ + d];
  __shared__ float s1[DE_], s2[DE_];
  s1[d] = v; s2[d] = v*v;
  __syncthreads();
  for (int s = 96; s > 2; s >>= 1){
    if (d < s){ s1[d] += s1[d+s]; s2[d] += s2[d+s]; }
    __syncthreads();
  }
  float sum = s1[0]+s1[1]+s1[2];
  float ssq = s2[0]+s2[1]+s2[2];
  float mu  = sum * (1.f/DE_);
  float var = ssq * (1.f/DE_) - mu*mu;
  float yn = (v - mu) * rsqrtf(var + 1e-5f) * gam[d] + bet[d];
  float z  = xz[(size_t)m*(2*DE_) + DE_ + d];
  ylnh[(size_t)m*DE_ + d] = (_Float16)(yn * (z * sigmoidf_(z)));
}

// ---------------- launch ----------------
extern "C" void kernel_launch(void* const* d_in, const int* in_sizes, int n_in,
                              void* d_out, int out_size, void* d_ws, size_t ws_size,
                              hipStream_t stream) {
  const float* x    = (const float*)d_in[0];
  const float* ipw  = (const float*)d_in[1];   // (384, 96)
  const float* cw   = (const float*)d_in[2];   // (192,1,3,3)
  const float* cb   = (const float*)d_in[3];   // (192,)
  const float* xpw  = (const float*)d_in[4];   // (2, 38, 192)
  const float* dtw  = (const float*)d_in[5];   // (2, 192, 6)
  const float* dtb  = (const float*)d_in[6];   // (2, 192)
  const float* alog = (const float*)d_in[7];   // (384, 16)
  const float* Dsv  = (const float*)d_in[8];   // (384,)
  const float* gam  = (const float*)d_in[9];
  const float* bet  = (const float*)d_in[10];
  const float* opw  = (const float*)d_in[11];  // (96, 192)
  float* out = (float*)d_out;

  char* ws = (char*)d_ws;
  _Float16* wA   = (_Float16*)(ws + 0);          // 384*96  f16
  _Float16* wB   = (_Float16*)(ws + 73728);      // 80*192  f16 (padded x_proj)
  _Float16* wC   = (_Float16*)(ws + 104448);     // 96*192  f16
  float*    xz   = (float*)(ws + 141312);        // 32768*384 f32
  float*    xc   = (float*)(ws + 50472960);      // 32768*192 f32
  float*    proj = (float*)(ws + 75638784);      // 32768*80  f32
  float*    ysb  = (float*)(ws + 86124544);      // 512*128*192 f32
  _Float16* xh   = (_Float16*)(ws + 136456192);  // 32768*96  f16
  _Float16* xch  = (_Float16*)(ws + 142747648);  // 32768*192 f16
  _Float16* ylnh = (_Float16*)(ws + 155330560);  // 32768*192 f16 (ends ~168 MB)

  // weight + activation converts
  k_cvt_f16   <<<(384*96  + 255)/256, 256, 0, stream>>>(ipw, wA, 384*96);
  k_pack_xproj<<<(80*DE_  + 255)/256, 256, 0, stream>>>(xpw, wB);
  k_cvt_f16   <<<(96*DE_  + 255)/256, 256, 0, stream>>>(opw, wC, 96*DE_);
  k_cvt_f16   <<<(MTOT*DM_+ 255)/256, 256, 0, stream>>>(x, xh, MTOT*DM_);

  // in_proj: xz = x @ in_proj_w^T   (M=32768, N=384, K=96)
  k_wmma_gemm<<<dim3(384/16, MTOT/64), 128, 0, stream>>>(xh, wA, xz, MTOT, 384, 96);
  // depthwise conv + SiLU (fp32 for scan, f16 for next GEMM)
  k_conv_dw<<<(MTOT*DE_ + 255)/256, 256, 0, stream>>>(xz, cw, cb, xc, xch);
  // x_dbl for both directions: proj = xc @ Wcat^T  (M=32768, N=80, K=192)
  k_wmma_gemm<<<dim3(80/16, MTOT/64), 128, 0, stream>>>(xch, wB, proj, MTOT, 80, 192);
  // bidirectional selective scan along H
  k_scan<<<B_*W_*2, 192, 0, stream>>>(xc, proj, dtw, dtb, alog, Dsv, ysb);
  // merge + LN + gate -> f16
  k_merge_ln<<<MTOT, 192, 0, stream>>>(ysb, xz, gam, bet, ylnh);
  // out_proj: out = yln @ out_proj_w^T  (M=32768, N=96, K=192)
  k_wmma_gemm<<<dim3(96/16, MTOT/64), 128, 0, stream>>>(ylnh, wC, out, MTOT, 96, 192);
}
